// CETLayer_37263136260545
// MI455X (gfx1250) — matrix operand order repaired and checked by reading, not compiled, not run on hardware
//
#include <hip/hip_runtime.h>

typedef __attribute__((ext_vector_type(16))) _Float16 v16h;
typedef __attribute__((ext_vector_type(8)))  float    v8f;

#define NNODES 20000
#define NEDGES 640000
#define DIM    128
#define NTYP   8
#define FLT_NEG_MAX (-3.402823466e38f)

// ---------------- zero init (CSR counters) ----------------
__global__ void zero_kernel(unsigned* __restrict__ p, int n) {
    int i = blockIdx.x * blockDim.x + threadIdx.x;
    if (i < n) p[i] = 0u;
}

// ---------------- CSR build: histogram ----------------
__global__ void hist_kernel(const int* __restrict__ dst, unsigned* __restrict__ cnt) {
    int e = blockIdx.x * blockDim.x + threadIdx.x;
    if (e < NEDGES) atomicAdd(&cnt[dst[e]], 1u);
}

// ---------------- CSR build: single-block exclusive scan ----------------
__global__ __launch_bounds__(1024) void scan_kernel(const unsigned* __restrict__ cnt,
                                                    unsigned* __restrict__ rowptr) {
    __shared__ unsigned part[1024];
    const int tid = threadIdx.x;
    const int CH  = 20;                      // 1024*20 >= 20000
    const int base = tid * CH;
    unsigned loc[CH];
    unsigned s = 0;
#pragma unroll
    for (int i = 0; i < CH; ++i) {
        int idx = base + i;
        unsigned v = (idx < NNODES) ? cnt[idx] : 0u;
        loc[i] = v;
        s += v;
    }
    part[tid] = s;
    __syncthreads();
    for (int off = 1; off < 1024; off <<= 1) {   // Hillis-Steele inclusive scan
        unsigned v = (tid >= off) ? part[tid - off] : 0u;
        __syncthreads();
        part[tid] += v;
        __syncthreads();
    }
    unsigned run = part[tid] - s;                // exclusive prefix for this chunk
#pragma unroll
    for (int i = 0; i < CH; ++i) {
        int idx = base + i;
        if (idx < NNODES) rowptr[idx] = run;
        run += loc[i];
    }
    if (tid == 0) rowptr[NNODES] = NEDGES;
}

// ---------------- CSR build: fill edge ids ----------------
__global__ void fill_kernel(const int* __restrict__ dst, const unsigned* __restrict__ rowptr,
                            unsigned* __restrict__ cursor, unsigned* __restrict__ eidx) {
    int e = blockIdx.x * blockDim.x + threadIdx.x;
    if (e < NEDGES) {
        int d = dst[e];
        unsigned pos = atomicAdd(&cursor[d], 1u);
        eidx[rowptr[d] + pos] = (unsigned)e;
    }
}

// Wave-level GEMM: A tile (16 rows x 128 relu'd f16) in rowp (stride-136 rows),
// W f16 tile (16x128, rows 8..15 zero). C is 16x16 f32 (cols N<8 meaningful).
__device__ __forceinline__ v8f wave_gemm_16x128x16(
    const _Float16* __restrict__ rowp,
    const _Float16 (* __restrict__ sW)[128],
    int lane)
{
    const int hi = (lane & 16) ? 1 : 0;
    const _Float16* wrow = &sW[lane & 15][0];
    v8f c = {};
#pragma unroll
    for (int kk = 0; kk < 4; ++kk) {
        const int k0    = kk * 32;
        const int abase = k0 + (hi ? 8 : 0);   // A: lanes>=16 hold K+8
        const int bbase = k0 + (hi ? 8 : 0);   // B: lanes>=16 hold K+8
        v16h a, b;
#pragma unroll
        for (int h = 0; h < 8; ++h) {
            a[h]     = rowp[abase + h];
            a[h + 8] = rowp[abase + 16 + h];
        }
#pragma unroll
        for (int h = 0; h < 8; ++h) {
            b[h]     = wrow[bbase + h];
            b[h + 8] = wrow[bbase + 16 + h];
        }
        c = __builtin_amdgcn_wmma_f32_16x16x32_f16(
                false, a, false, b, (short)0, c, false, false);
    }
    return c;
}

// ---------------- edge pass: p1 = relu(msg) @ W^T + b (no atomics) ----------------
// 8 waves/block, 16 edges/wave, grid = E/128 = 5000
__global__ __launch_bounds__(256) void edge_p1(
    const float* __restrict__ srcEmb, const float* __restrict__ edgeEmb,
    const float* __restrict__ W, const float* __restrict__ bias,
    const int* __restrict__ src,
    float* __restrict__ p1out)
{
    __shared__ _Float16 sW[16][128];
    __shared__ float    sB[16];
    __shared__ _Float16 sMsg[8][16][136];

    const int tid  = threadIdx.x;
    const int wv   = tid >> 5;
    const int lane = tid & 31;

    for (int i = tid; i < 16 * 128; i += 256) {
        int r = i >> 7, k = i & 127;
        sW[r][k] = (r < NTYP) ? (_Float16)W[r * DIM + k] : (_Float16)0.f;
    }
    if (tid < 16) sB[tid] = (tid < NTYP) ? bias[tid] : 0.f;

    const int e0 = blockIdx.x * 128 + wv * 16;
    // prefetch the wave's 8KB edge tile (gfx1250 global_prefetch_b8)
    __builtin_prefetch(edgeEmb + (size_t)e0 * DIM + (size_t)lane * 128, 0, 1);
    __syncthreads();

#pragma unroll 4
    for (int el = 0; el < 16; ++el) {
        const int e = e0 + el;
        const int s = src[e];
        const float4 ev = *(const float4*)(edgeEmb + (size_t)e * DIM + lane * 4);
        const float4 sv = *(const float4*)(srcEmb  + (size_t)s * DIM + lane * 4);
        _Float16* mr = &sMsg[wv][el][lane * 4];
        mr[0] = (_Float16)fmaxf(ev.x + sv.x, 0.f);
        mr[1] = (_Float16)fmaxf(ev.y + sv.y, 0.f);
        mr[2] = (_Float16)fmaxf(ev.z + sv.z, 0.f);
        mr[3] = (_Float16)fmaxf(ev.w + sv.w, 0.f);
    }
    __syncthreads();

    v8f c = wave_gemm_16x128x16(&sMsg[wv][lane & 15][0], sW, lane);

    const int hi = (lane & 16) ? 1 : 0;
    const int t  = lane & 15;
    const float bv = sB[t];
#pragma unroll
    for (int r = 0; r < 8; ++r) {
        const int M = r + (hi ? 8 : 0);
        if (t < NTYP)
            p1out[(size_t)(e0 + M) * NTYP + t] = c[r] + bv;
    }
}

// ---------------- gather pass: agg sums + segment max of p1 (no atomics) ----------
// one wave per node; 8 waves/block; grid = N/8 = 2500
__global__ __launch_bounds__(256) void gather_agg(
    const float* __restrict__ srcEmb, const float* __restrict__ edgeEmb,
    const int* __restrict__ src,
    const unsigned* __restrict__ rowptr, const unsigned* __restrict__ eidx,
    const float* __restrict__ p1,
    float* __restrict__ agg, float* __restrict__ m1f)
{
    const int tid  = threadIdx.x;
    const int wv   = tid >> 5;
    const int lane = tid & 31;
    const int n    = blockIdx.x * 8 + wv;

    const unsigned beg = rowptr[n];
    const unsigned end = rowptr[n + 1];

    float a0 = 0.f, a1 = 0.f, a2 = 0.f, a3 = 0.f;
    float pmax = FLT_NEG_MAX;
    for (unsigned j = beg; j < end; ++j) {
        const int e = (int)eidx[j];
        const int s = src[e];
        const float4 ev = *(const float4*)(edgeEmb + (size_t)e * DIM + lane * 4);
        const float4 sv = *(const float4*)(srcEmb  + (size_t)s * DIM + lane * 4);
        a0 += ev.x + sv.x; a1 += ev.y + sv.y;
        a2 += ev.z + sv.z; a3 += ev.w + sv.w;
        if (lane < NTYP) pmax = fmaxf(pmax, p1[(size_t)e * NTYP + lane]);
    }
    float4 o; o.x = a0; o.y = a1; o.z = a2; o.w = a3;
    *(float4*)(agg + (size_t)n * DIM + lane * 4) = o;
    if (lane < NTYP) m1f[(size_t)n * NTYP + lane] = pmax;
}

// ---------------- node pass: p2 GEMM + final max ----------------
__global__ __launch_bounds__(256) void node_pass(
    const float* __restrict__ W, const float* __restrict__ bias,
    const float* __restrict__ agg, const unsigned* __restrict__ rowptr,
    const float* __restrict__ m1f,
    float* __restrict__ p2out, float* __restrict__ mout)
{
    __shared__ _Float16 sW[16][128];
    __shared__ float    sB[16];
    __shared__ _Float16 sMsg[8][16][136];

    const int tid  = threadIdx.x;
    const int wv   = tid >> 5;
    const int lane = tid & 31;

    for (int i = tid; i < 16 * 128; i += 256) {
        int r = i >> 7, k = i & 127;
        sW[r][k] = (r < NTYP) ? (_Float16)W[r * DIM + k] : (_Float16)0.f;
    }
    if (tid < 16) sB[tid] = (tid < NTYP) ? bias[tid] : 0.f;
    __syncthreads();

    const int n0 = blockIdx.x * 128 + wv * 16;

#pragma unroll 4
    for (int nl = 0; nl < 16; ++nl) {
        const int n  = n0 + nl;
        const int nc = (n < NNODES) ? n : (NNODES - 1);
        const float dg  = (float)(rowptr[nc + 1] - rowptr[nc]);
        const float inv = 1.0f / fmaxf(dg, 1.0f);
        const float4 av = *(const float4*)(agg + (size_t)nc * DIM + lane * 4);
        _Float16* mr = &sMsg[wv][nl][lane * 4];
        mr[0] = (_Float16)fmaxf(av.x * inv, 0.f);
        mr[1] = (_Float16)fmaxf(av.y * inv, 0.f);
        mr[2] = (_Float16)fmaxf(av.z * inv, 0.f);
        mr[3] = (_Float16)fmaxf(av.w * inv, 0.f);
    }
    __syncthreads();

    v8f c = wave_gemm_16x128x16(&sMsg[wv][lane & 15][0], sW, lane);

    const int hi = (lane & 16) ? 1 : 0;
    const int t  = lane & 15;
    const float bv = sB[t];
#pragma unroll
    for (int r = 0; r < 8; ++r) {
        const int M = r + (hi ? 8 : 0);
        const int n = n0 + M;
        if (t < NTYP && n < NNODES) {
            const float p2v = c[r] + bv;
            p2out[(size_t)n * NTYP + t] = p2v;
            mout [(size_t)n * NTYP + t] = fmaxf(m1f[(size_t)n * NTYP + t], p2v);
        }
    }
}

// ---------------- fused softmax + sigmoid (gather, no atomics) ----------------
// 8 threads per node (thread t of group handles type t); grid = N*8/256 = 625
__global__ __launch_bounds__(256) void node_softmax(
    const float* __restrict__ p1, const float* __restrict__ p2ws,
    const float* __restrict__ mws,
    const unsigned* __restrict__ rowptr, const unsigned* __restrict__ eidx,
    float* __restrict__ out)
{
    const int g = blockIdx.x * blockDim.x + threadIdx.x;
    const int n = g >> 3;
    const int t = g & 7;
    const unsigned beg = rowptr[n];
    const unsigned end = rowptr[n + 1];
    const float mm  = mws[g];
    const float p2v = p2ws[g];
    float den = 0.f, num = 0.f;
    for (unsigned j = beg; j < end; ++j) {
        const int   e  = (int)eidx[j];
        const float p  = p1[(size_t)e * NTYP + t];
        const float e1 = __expf(p - mm);
        den += e1;
        num += p * e1;
    }
    const float e2 = __expf(p2v - mm);
    den += e2;
    num += p2v * e2;
    const float s = num / den;
    out[g] = (end > beg) ? 1.0f / (1.0f + __expf(-s)) : 0.0f;
}

extern "C" void kernel_launch(void* const* d_in, const int* in_sizes, int n_in,
                              void* d_out, int out_size, void* d_ws, size_t ws_size,
                              hipStream_t stream) {
    const float* srcEmb  = (const float*)d_in[0];  // [N,128]
    const float* edgeEmb = (const float*)d_in[1];  // [E,128]
    const float* W       = (const float*)d_in[2];  // [8,128]
    const float* bias    = (const float*)d_in[3];  // [8]
    const int*   src     = (const int*)d_in[4];    // [E]
    const int*   dst     = (const int*)d_in[5];    // [E]
    float*       out     = (float*)d_out;          // [N,8]

    // ws layout (all chunk sizes multiples of 8 elements -> 16B alignment kept)
    unsigned* cnt    = (unsigned*)d_ws;                       // N
    unsigned* cursor = cnt + NNODES;                          // N
    unsigned* rowptr = cursor + NNODES;                       // N+1 (padded to N+8)
    unsigned* eidx   = rowptr + (NNODES + 8);                 // E
    float*    p1     = (float*)(eidx + NEDGES);               // E*8
    float*    agg    = p1 + (size_t)NEDGES * NTYP;            // N*128
    float*    m1f    = agg + (size_t)NNODES * DIM;            // N*8
    float*    p2     = m1f + (size_t)NNODES * NTYP;           // N*8
    float*    mfin   = p2 + (size_t)NNODES * NTYP;            // N*8

    // CSR build
    zero_kernel<<<(2 * NNODES + 255) / 256, 256, 0, stream>>>(cnt, 2 * NNODES);
    hist_kernel<<<(NEDGES + 255) / 256, 256, 0, stream>>>(dst, cnt);
    scan_kernel<<<1, 1024, 0, stream>>>(cnt, rowptr);
    fill_kernel<<<(NEDGES + 255) / 256, 256, 0, stream>>>(dst, rowptr, cursor, eidx);

    // main passes
    edge_p1<<<NEDGES / 128, 256, 0, stream>>>(srcEmb, edgeEmb, W, bias, src, p1);
    gather_agg<<<NNODES / 8, 256, 0, stream>>>(srcEmb, edgeEmb, src, rowptr, eidx,
                                               p1, agg, m1f);
    node_pass<<<(NNODES + 127) / 128, 256, 0, stream>>>(W, bias, agg, rowptr, m1f,
                                                        p2, mfin);
    node_softmax<<<(NNODES * NTYP) / 256, 256, 0, stream>>>(p1, p2, mfin, rowptr,
                                                            eidx, out);
}
